// MultiHeadAttention_58523224375801
// MI455X (gfx1250) — compile-verified, matrix operand order checked by reading, not run on hardware
//
#include <hip/hip_runtime.h>
#include <math.h>

#define DIM    768
#define NHEAD  16
#define HD     48
#define HDP    64
#define SEQ    1024
#define BATCH  8
#define NTOK   (BATCH*SEQ)      /* 8192 */

typedef __bf16 bf16;
typedef __attribute__((ext_vector_type(16))) __bf16 v16bf;
typedef __attribute__((ext_vector_type(8)))  __bf16 v8bf;
typedef __attribute__((ext_vector_type(8)))  float  v8f;

// Load one 16x32 bf16 WMMA operand fragment (A-layout; identical striping is used
// for B when the source is stored transposed, i.e. Bt[N][K] row-major).
// ISA 16-bit operand layout: lanes 0-15 hold row = lane, K = {0..7,16..23};
// lanes 16-31 hold row = lane-16, K = {8..15,24..31}.  Two 16-byte loads/lane.
__device__ __forceinline__ v16bf load_frag(const bf16* __restrict__ base, int ld, int k0) {
  const int lane = threadIdx.x & 31;
  const int row  = lane & 15;
  const int kb   = (lane & 16) ? 8 : 0;
  const bf16* p = base + row * ld + k0 + kb;
  v8bf lo = *(const v8bf*)p;
  v8bf hi = *(const v8bf*)(p + 16);
  v16bf f;
#pragma unroll
  for (int e = 0; e < 8; ++e) { f[e] = lo[e]; f[e + 8] = hi[e]; }
  return f;
}

__global__ void k_cvt_bf16(const float* __restrict__ src, bf16* __restrict__ dst, int n) {
  int i = blockIdx.x * blockDim.x + threadIdx.x;
  if (i < n) dst[i] = (bf16)src[i];
}

__global__ void k_transpose_bf16(const float* __restrict__ w, bf16* __restrict__ wT) {
  int i = blockIdx.x * blockDim.x + threadIdx.x;
  if (i < DIM * DIM) {
    int n = i / DIM, k = i - n * DIM;
    wT[i] = (bf16)w[k * DIM + n];   // wT[n][k] = w[k][n]
  }
}

__global__ void k_zero_bf16(bf16* __restrict__ dst, int n) {
  int i = blockIdx.x * blockDim.x + threadIdx.x;
  if (i < n) dst[i] = (bf16)0.0f;
}

// Epilogue for one 16x16 f32 accumulator tile.
// mode 0: f32 row-major [M x 768] -> ofp (final projection output)
// mode 1: bf16, (val+bias)*scale -> obf[((b*H+h)*SEQ+s)*HDP + d]   (Q/K, d<48)
// mode 2: bf16, (val+bias)       -> obf[((b*H+h)*HD+d)*SEQ + s]    (V transposed)
__device__ __forceinline__ void store_tile(v8f acc, int m0, int n0,
                                           const float* __restrict__ bias,
                                           int mode, float scale,
                                           bf16* __restrict__ obf,
                                           float* __restrict__ ofp) {
  const int lane = threadIdx.x & 31;
  const int n  = n0 + (lane & 15);
  const int mh = (lane & 16) ? 8 : 0;
  const float bval = bias[n];
  const int h = n / HD, d = n - h * HD;   // 48 = 3*16 -> a tile never straddles heads
#pragma unroll
  for (int r = 0; r < 8; ++r) {
    const int m = m0 + mh + r;
    const float val = acc[r] + bval;
    if (mode == 0) {
      ofp[(size_t)m * DIM + n] = val;
    } else {
      const int b = m >> 10, s = m & (SEQ - 1);
      if (mode == 1)
        obf[(((size_t)(b * NHEAD + h) * SEQ + s) * HDP) + d] = (bf16)(val * scale);
      else
        obf[(((size_t)(b * NHEAD + h) * HD + d) * SEQ) + s] = (bf16)val;
    }
  }
}

// C = A[8192x768] * Bt^T + bias, register-blocked: each wave owns a 64x32 block
// (4 M-tiles x 2 N-tiles -> 8 wmma per 6 fragment loads).  8 waves per block
// arranged 2(M) x 4(N) -> 128x128 per workgroup.  Grid: (8192/128)*(768/128)=384.
__global__ void k_gemm(const bf16* __restrict__ A, const bf16* __restrict__ Bt,
                       const float* __restrict__ bias, int mode, float scale,
                       bf16* __restrict__ obf, float* __restrict__ ofp) {
  const int w  = threadIdx.x >> 5;          // 0..7
  const int wm = w >> 2, wn = w & 3;        // 2 x 4 wave grid
  const int bm = blockIdx.x / (DIM / 128);  // 0..63
  const int bn = blockIdx.x % (DIM / 128);  // 0..5
  const int m0 = bm * 128 + wm * 64;
  const int n0 = bn * 128 + wn * 32;
  const bf16* Ap = A  + (size_t)m0 * DIM;
  const bf16* Bp = Bt + (size_t)n0 * DIM;

  v8f acc[4][2];
#pragma unroll
  for (int i = 0; i < 4; ++i)
#pragma unroll
    for (int j = 0; j < 2; ++j) acc[i][j] = (v8f){};

  for (int k0 = 0; k0 < DIM; k0 += 32) {
    v16bf fa[4], fb[2];
#pragma unroll
    for (int i = 0; i < 4; ++i) fa[i] = load_frag(Ap + (size_t)(i * 16) * DIM, DIM, k0);
#pragma unroll
    for (int j = 0; j < 2; ++j) fb[j] = load_frag(Bp + (size_t)(j * 16) * DIM, DIM, k0);
#pragma unroll
    for (int i = 0; i < 4; ++i)
#pragma unroll
      for (int j = 0; j < 2; ++j)
        acc[i][j] = __builtin_amdgcn_wmma_f32_16x16x32_bf16(false, fa[i], false, fb[j],
                                                            (short)0, acc[i][j], false, false);
  }
#pragma unroll
  for (int i = 0; i < 4; ++i)
#pragma unroll
    for (int j = 0; j < 2; ++j)
      store_tile(acc[i][j], m0 + i * 16, n0 + j * 16, bias, mode, scale, obf, ofp);
}

// Flash attention, one wave per (b, h, 32-query block) = 2 query tiles per wave,
// so every K/V fragment is reused twice: 14 wmma per 7 fragment loads / key-step.
// Computes S^T = K * Q^T: the score accumulator layout IS the B-operand layout of
// P^T for O^T = V^T * P^T -> no cross-lane transpose; softmax stats need a single
// shfl_xor(16).  Q pre-scaled by 1/sqrt(48); q/k zero-padded to d=64.
__global__ void k_attn(const bf16* __restrict__ qh, const bf16* __restrict__ kh,
                       const bf16* __restrict__ vT, bf16* __restrict__ ctx) {
  const int wave = (blockIdx.x * blockDim.x + threadIdx.x) >> 5;
  const int q2 = wave & 31;          // 32 double-query-blocks of 32 queries
  const int bh = wave >> 5;          // 0..127
  const int h = bh & (NHEAD - 1), b = bh >> 4;
  const int lane = threadIdx.x & 31;

  const bf16* qp = qh + (size_t)bh * SEQ * HDP;
  const bf16* kp = kh + (size_t)bh * SEQ * HDP;
  const bf16* vp = vT + (size_t)bh * HD * SEQ;

  v16bf qf[2][2];
#pragma unroll
  for (int g = 0; g < 2; ++g) {
    const bf16* qg = qp + (size_t)(q2 * 32 + g * 16) * HDP;
    qf[g][0] = load_frag(qg, HDP, 0);
    qf[g][1] = load_frag(qg, HDP, 32);
  }

  float mrun[2] = {-INFINITY, -INFINITY};
  float lrun[2] = {0.0f, 0.0f};
  v8f o[2][3];
#pragma unroll
  for (int g = 0; g < 2; ++g)
#pragma unroll
    for (int t = 0; t < 3; ++t) o[g][t] = (v8f){};

  for (int kk = 0; kk < SEQ; kk += 32) {
    const v16bf ka0 = load_frag(kp + (size_t)kk * HDP, HDP, 0);
    const v16bf ka1 = load_frag(kp + (size_t)kk * HDP, HDP, 32);
    const v16bf kb0 = load_frag(kp + (size_t)(kk + 16) * HDP, HDP, 0);
    const v16bf kb1 = load_frag(kp + (size_t)(kk + 16) * HDP, HDP, 32);

    v16bf fpv[2];
#pragma unroll
    for (int g = 0; g < 2; ++g) {
      v8f s0 = {}, s1 = {};
      s0 = __builtin_amdgcn_wmma_f32_16x16x32_bf16(false, ka0, false, qf[g][0], (short)0, s0, false, false);
      s0 = __builtin_amdgcn_wmma_f32_16x16x32_bf16(false, ka1, false, qf[g][1], (short)0, s0, false, false);
      s1 = __builtin_amdgcn_wmma_f32_16x16x32_bf16(false, kb0, false, qf[g][0], (short)0, s1, false, false);
      s1 = __builtin_amdgcn_wmma_f32_16x16x32_bf16(false, kb1, false, qf[g][1], (short)0, s1, false, false);

      // online softmax over this 32-key block (row = query = lane%16)
      float tmax = -INFINITY;
#pragma unroll
      for (int r = 0; r < 8; ++r) tmax = fmaxf(tmax, fmaxf(s0[r], s1[r]));
      tmax = fmaxf(tmax, __shfl_xor(tmax, 16, 32));
      const float mnew = fmaxf(mrun[g], tmax);
      const float corr = __expf(mrun[g] - mnew);
      float p0[8], p1[8], psum = 0.0f;
#pragma unroll
      for (int r = 0; r < 8; ++r) {
        p0[r] = __expf(s0[r] - mnew);
        p1[r] = __expf(s1[r] - mnew);
        psum += p0[r] + p1[r];
      }
      psum += __shfl_xor(psum, 16, 32);
      lrun[g] = lrun[g] * corr + psum;
      mrun[g] = mnew;
#pragma unroll
      for (int t = 0; t < 3; ++t)
#pragma unroll
        for (int r = 0; r < 8; ++r) o[g][t][r] *= corr;
      // pack P^T B-fragment: element e<8 -> tile0 vgpr e, e>=8 -> tile1 vgpr e-8
#pragma unroll
      for (int e = 0; e < 8; ++e) { fpv[g][e] = (bf16)p0[e]; fpv[g][e + 8] = (bf16)p1[e]; }
    }

    // O^T += V^T * P^T  (3 d-tiles of 16; V^T rows contiguous over keys),
    // each V fragment reused for both query tiles.
#pragma unroll
    for (int t = 0; t < 3; ++t) {
      const v16bf vf = load_frag(vp + (size_t)(t * 16) * SEQ, SEQ, kk);
      o[0][t] = __builtin_amdgcn_wmma_f32_16x16x32_bf16(false, vf, false, fpv[0], (short)0, o[0][t], false, false);
      o[1][t] = __builtin_amdgcn_wmma_f32_16x16x32_bf16(false, vf, false, fpv[1], (short)0, o[1][t], false, false);
    }
  }

  const int dh = (lane & 16) ? 8 : 0;
#pragma unroll
  for (int g = 0; g < 2; ++g) {
    const float inv = 1.0f / lrun[g];
    const int q = q2 * 32 + g * 16 + (lane & 15);
    bf16* cp = ctx + ((size_t)(b * SEQ + q)) * DIM + h * HD;
#pragma unroll
    for (int r = 0; r < 8; ++r) {
      cp[ 0 + dh + r] = (bf16)(o[g][0][r] * inv);
      cp[16 + dh + r] = (bf16)(o[g][1][r] * inv);
      cp[32 + dh + r] = (bf16)(o[g][2][r] * inv);
    }
  }
}

extern "C" void kernel_launch(void* const* d_in, const int* in_sizes, int n_in,
                              void* d_out, int out_size, void* d_ws, size_t ws_size,
                              hipStream_t stream) {
  const float* x  = (const float*)d_in[0];
  const float* wq = (const float*)d_in[1];
  const float* bq = (const float*)d_in[2];
  const float* wk = (const float*)d_in[3];
  const float* bk = (const float*)d_in[4];
  const float* wv = (const float*)d_in[5];
  const float* bv = (const float*)d_in[6];
  const float* wo = (const float*)d_in[7];
  const float* bo = (const float*)d_in[8];
  float* out = (float*)d_out;

  char* ws = (char*)d_ws;
  size_t off = 0;
  auto alloc = [&](size_t bytes) -> void* {
    void* p = ws + off;
    off += (bytes + 255) & ~(size_t)255;
    return p;
  };
  bf16* xbf = (bf16*)alloc((size_t)NTOK * DIM * 2);                 // 12.6 MB
  bf16* wqT = (bf16*)alloc((size_t)DIM * DIM * 2);
  bf16* wkT = (bf16*)alloc((size_t)DIM * DIM * 2);
  bf16* wvT = (bf16*)alloc((size_t)DIM * DIM * 2);
  bf16* woT = (bf16*)alloc((size_t)DIM * DIM * 2);                  // 4 x 1.2 MB
  bf16* qhB = (bf16*)alloc((size_t)BATCH * NHEAD * SEQ * HDP * 2);  // 16.8 MB
  bf16* khB = (bf16*)alloc((size_t)BATCH * NHEAD * SEQ * HDP * 2);  // 16.8 MB
  bf16* vTB = (bf16*)alloc((size_t)BATCH * NHEAD * HD * SEQ * 2);   // 12.6 MB
  bf16* ctx = (bf16*)alloc((size_t)NTOK * DIM * 2);                 // 12.6 MB
  (void)ws_size; (void)in_sizes; (void)n_in; (void)out_size;

  const int nx = NTOK * DIM;
  const int nw = DIM * DIM;
  const int nqk = BATCH * NHEAD * SEQ * HDP;

  k_cvt_bf16<<<(nx + 255) / 256, 256, 0, stream>>>(x, xbf, nx);
  k_transpose_bf16<<<(nw + 255) / 256, 256, 0, stream>>>(wq, wqT);
  k_transpose_bf16<<<(nw + 255) / 256, 256, 0, stream>>>(wk, wkT);
  k_transpose_bf16<<<(nw + 255) / 256, 256, 0, stream>>>(wv, wvT);
  k_transpose_bf16<<<(nw + 255) / 256, 256, 0, stream>>>(wo, woT);
  k_zero_bf16<<<(nqk + 255) / 256, 256, 0, stream>>>(qhB, nqk);
  k_zero_bf16<<<(nqk + 255) / 256, 256, 0, stream>>>(khB, nqk);

  // 8 waves/block, each wave computes 64x32; block covers 128x128.
  const int gemm_blocks = (NTOK / 128) * (DIM / 128);   // 64*6 = 384
  const float qscale = 1.0f / sqrtf((float)HD);
  k_gemm<<<gemm_blocks, 256, 0, stream>>>(xbf, wqT, bq, 1, qscale, qhB, nullptr);
  k_gemm<<<gemm_blocks, 256, 0, stream>>>(xbf, wkT, bk, 1, 1.0f,  khB, nullptr);
  k_gemm<<<gemm_blocks, 256, 0, stream>>>(xbf, wvT, bv, 2, 1.0f,  vTB, nullptr);

  // one wave per 32 queries: 8*16*32 = 4096 waves / 8 per block = 512 blocks
  const int attn_blocks = (BATCH * NHEAD * (SEQ / 32)) / 8;
  k_attn<<<attn_blocks, 256, 0, stream>>>(qhB, khB, vTB, ctx);

  k_gemm<<<gemm_blocks, 256, 0, stream>>>(ctx, woT, bo, 0, 1.0f, nullptr, out);
}
